// HybridTransformerBlock_65481071396451
// MI455X (gfx1250) — matrix operand order repaired and checked by reading, not compiled, not run on hardware
//
#include <hip/hip_runtime.h>

// ---------------------------------------------------------------------------
// Types for CDNA5 WMMA (wave32, v_wmma_f32_16x16x32_bf16)
// ---------------------------------------------------------------------------
typedef __bf16 bf16;
typedef __attribute__((ext_vector_type(16))) __bf16 v16bf;
typedef __attribute__((ext_vector_type(8)))  __bf16 v8bf;
typedef __attribute__((ext_vector_type(8)))  float  v8f;
typedef unsigned int u32x4 __attribute__((ext_vector_type(4)));
typedef int          i32x8 __attribute__((ext_vector_type(8)));
typedef int          i32x4 __attribute__((ext_vector_type(4)));

// Tensor Data Mover availability (guarded so a missing builtin can't break
// the build; host pass sees 0 and parses only the fallback path).
#if defined(__has_builtin)
#if __has_builtin(__builtin_amdgcn_tensor_load_to_lds)
#define USE_TDM 1
#endif
#endif
#ifndef USE_TDM
#define USE_TDM 0
#endif

__device__ __forceinline__ v8f wmma_bf16(v16bf a, v16bf b, v8f c) {
  // (neg_a, A, neg_b, B, c_mod, C, reuse_a, reuse_b)
  return __builtin_amdgcn_wmma_f32_16x16x32_bf16(false, a, false, b, (short)0, c,
                                                 false, false);
}

// A fragment (16x32 bf16), ISA 7.12.2 layout:
//   lanes 0-15 : row = lane,    elems 0..7 = K 0..7,  elems 8..15 = K 16..23
//   lanes16-31 : row = lane-16, elems 0..7 = K 8..15, elems 8..15 = K 24..31
__device__ __forceinline__ v16bf load_a_frag(const bf16* base, int stride, int lane) {
  const bf16* p = base + (lane & 15) * stride + ((lane >> 4) << 3);
  v8bf lo = *(const v8bf*)(p);
  v8bf hi = *(const v8bf*)(p + 16);
  return __builtin_shufflevector(lo, hi, 0, 1, 2, 3, 4, 5, 6, 7,
                                 8, 9, 10, 11, 12, 13, 14, 15);
}

// B fragment (32x16 bf16):
//   lanes 0-15 : col = lane,    elems 0..7 = K 0..7,  elems 8..15 = K 16..23
//   lanes16-31 : col = lane-16, elems 0..7 = K 8..15, elems 8..15 = K 24..31
// LDS tile is stored column-major for B: Bs[col][k], k contiguous.
__device__ __forceinline__ v16bf load_b_frag(const bf16* base, int stride, int lane) {
  const bf16* p = base + (lane & 15) * stride + ((lane >> 4) << 3);
  v8bf lo = *(const v8bf*)(p);
  v8bf hi = *(const v8bf*)(p + 16);
  return __builtin_shufflevector(lo, hi, 0, 1, 2, 3, 4, 5, 6, 7,
                                 8, 9, 10, 11, 12, 13, 14, 15);
}

#if USE_TDM
// Issue one TDM 2D tile load: tile_rows rows of tile_k contiguous bf16
// elements, source row stride = stride_elems, into LDS at lds_byte_off.
// D# bit layout per CDNA5 ISA ch.8 (group0 128b, group1 256b).
__device__ __forceinline__ void tdm_load_tile(unsigned lds_byte_off,
                                              const bf16* gaddr, int tile_k,
                                              int tile_rows, int stride_elems) {
  unsigned long long ga = (unsigned long long)(__UINTPTR_TYPE__)gaddr;
  u32x4 g0;
  g0[0] = 1u;                                             // count=1 (valid user D#)
  g0[1] = lds_byte_off;                                   // lds_addr [63:32]
  g0[2] = (unsigned)(ga & 0xffffffffu);                   // global_addr lo
  g0[3] = (unsigned)((ga >> 32) & 0x01ffffffu) | (2u << 30);  // addr hi | type=2

  i32x8 g1;
  g1[0] = 0x00010000;                                     // data_size=1 (2 bytes)
  g1[1] = (int)((unsigned)(tile_k & 0xffff) << 16);       // tensor_dim0[15:0]
  g1[2] = (int)(((unsigned)tile_k >> 16) |
                ((unsigned)(tile_rows & 0xffff) << 16));  // dim0 hi | tensor_dim1 lo
  g1[3] = (int)(((unsigned)tile_rows >> 16) |
                ((unsigned)(tile_k & 0xffff) << 16));     // dim1 hi | tile_dim0
  g1[4] = (int)((unsigned)(tile_rows & 0xffff));          // tile_dim1 (tile_dim2=0)
  g1[5] = stride_elems;                                   // tensor_dim0_stride lo32
  g1[6] = 0;                                              // stride hi | dim1_stride
  g1[7] = 0;
  i32x4 z4 = {0, 0, 0, 0};
#if __clang_major__ >= 23
  i32x8 z8 = {0, 0, 0, 0, 0, 0, 0, 0};
  __builtin_amdgcn_tensor_load_to_lds(g0, g1, z4, z4, z8, 0);
#else
  __builtin_amdgcn_tensor_load_to_lds(g0, g1, z4, z4, 0);
#endif
}
#endif

// ---------------------------------------------------------------------------
// fp32 -> bf16 convert (activations)
// ---------------------------------------------------------------------------
__global__ __launch_bounds__(256) void convert_f32_bf16(const float* __restrict__ in,
                                                        bf16* __restrict__ out, int n) {
  int i = (blockIdx.x * 256 + threadIdx.x) * 4;
  if (i < n) {
    float4 f = *(const float4*)(in + i);
    out[i + 0] = (bf16)f.x;
    out[i + 1] = (bf16)f.y;
    out[i + 2] = (bf16)f.z;
    out[i + 3] = (bf16)f.w;
  }
}

// fp32 [K,N] -> bf16 transposed [N,K] (weights; makes GEMM B tiles k-contiguous
// so TDM / cooperative staging and B-fragment loads are all contiguous).
__global__ __launch_bounds__(256) void transpose_convert_f32_bf16(
    const float* __restrict__ in, bf16* __restrict__ out, int K, int N) {
  __shared__ float t[32][33];
  const int k0 = blockIdx.y * 32, n0 = blockIdx.x * 32;
  const int tx = threadIdx.x & 31, ty = threadIdx.x >> 5;  // 32 x 8
#pragma unroll
  for (int j = ty; j < 32; j += 8)
    t[j][tx] = in[(size_t)(k0 + j) * N + n0 + tx];
  __syncthreads();
#pragma unroll
  for (int j = ty; j < 32; j += 8)
    out[(size_t)(n0 + j) * K + k0 + tx] = (bf16)t[tx][j];
}

// ---------------------------------------------------------------------------
// Tiled bf16 WMMA GEMM: C[M,N] = epilogue(A[M,K] @ Bt[N,K]^T)
// Block tile 128x128; 8 waves, 32(M)x64(N) each; K-step 32.
// A/B tiles staged to double-buffered LDS by the Tensor Data Mover (wave 0
// issues descriptors, s_wait_tensorcnt 2 overlaps next-buffer DMA w/ compute).
// ---------------------------------------------------------------------------
__device__ __forceinline__ void gemm_compute_step(const bf16* As, const bf16* Bs,
                                                  v8f (&acc)[2][4], int wm, int wn,
                                                  int lane) {
  v16bf a0 = load_a_frag(As + (wm * 32 + 0)  * 32, 32, lane);
  v16bf a1 = load_a_frag(As + (wm * 32 + 16) * 32, 32, lane);
#pragma unroll
  for (int nt = 0; nt < 4; ++nt) {
    v16bf b = load_b_frag(Bs + (wn * 64 + nt * 16) * 32, 32, lane);
    acc[0][nt] = wmma_bf16(a0, b, acc[0][nt]);
    acc[1][nt] = wmma_bf16(a1, b, acc[1][nt]);
  }
}

template <bool OUT_BF16, bool HAS_BIAS, bool RELU, bool GATED>
__global__ __launch_bounds__(256) void gemm_bf16_wmma(
    const bf16* __restrict__ A, const bf16* __restrict__ Bt, void* __restrict__ Cout,
    const float* __restrict__ bias, const float* __restrict__ theta,
    int M, int N, int K) {
  __shared__ bf16 As[2][128 * 32];  // [m][k], k contiguous
  __shared__ bf16 Bs[2][128 * 32];  // [n][k], k contiguous

  const int tid  = threadIdx.x;
  const int lane = tid & 31;
  const int wid  = tid >> 5;
  const int wm   = wid >> 1;  // 4 waves along M (32 rows each)
  const int wn   = wid & 1;   // 2 waves along N (64 cols each)
  const int mblk = blockIdx.y * 128;
  const int nblk = blockIdx.x * 128;
  const int ktiles = K >> 5;

  v8f acc[2][4] = {};

#if USE_TDM
  if (wid == 0) {
    tdm_load_tile((unsigned)(__UINTPTR_TYPE__)&As[0][0], A + (size_t)mblk * K,
                  32, 128, K);
    tdm_load_tile((unsigned)(__UINTPTR_TYPE__)&Bs[0][0], Bt + (size_t)nblk * K,
                  32, 128, K);
  }
  for (int i = 0; i < ktiles; ++i) {
    const int cur = i & 1;
    if (wid == 0) {
      if (i + 1 < ktiles) {
        const int k0n = (i + 1) << 5;
        tdm_load_tile((unsigned)(__UINTPTR_TYPE__)&As[cur ^ 1][0],
                      A + (size_t)mblk * K + k0n, 32, 128, K);
        tdm_load_tile((unsigned)(__UINTPTR_TYPE__)&Bs[cur ^ 1][0],
                      Bt + (size_t)nblk * K + k0n, 32, 128, K);
        __builtin_amdgcn_s_wait_tensorcnt(2);  // oldest 2 (= cur buffer) done
      } else {
        __builtin_amdgcn_s_wait_tensorcnt(0);
      }
    }
    __syncthreads();  // publish cur buffer to all waves
    gemm_compute_step(As[cur], Bs[cur], acc, wm, wn, lane);
    __syncthreads();  // all waves done with the buffer TDM writes next
  }
#else
  for (int i = 0; i < ktiles; ++i) {
    const int k0 = i << 5;
    __syncthreads();
    {  // A tile 128x32: 16 contiguous elems per thread
      int r = tid >> 1, kh = (tid & 1) << 4;
      const bf16* g = A + (size_t)(mblk + r) * K + k0 + kh;
      *(v8bf*)(&As[0][0] + r * 32 + kh)     = *(const v8bf*)(g);
      *(v8bf*)(&As[0][0] + r * 32 + kh + 8) = *(const v8bf*)(g + 8);
    }
    {  // B tile 128x32 from Bt[N,K]: contiguous both sides
      int n = tid >> 1, kh = (tid & 1) << 4;
      const bf16* g = Bt + (size_t)(nblk + n) * K + k0 + kh;
      *(v8bf*)(&Bs[0][0] + n * 32 + kh)     = *(const v8bf*)(g);
      *(v8bf*)(&Bs[0][0] + n * 32 + kh + 8) = *(const v8bf*)(g + 8);
    }
    __syncthreads();
    gemm_compute_step(As[0], Bs[0], acc, wm, wn, lane);
  }
#endif

  float gate = 1.0f;
  if (GATED) gate = 1.0f / (1.0f + __expf(-__cosf(theta[0])));

#pragma unroll
  for (int mt = 0; mt < 2; ++mt) {
#pragma unroll
    for (int nt = 0; nt < 4; ++nt) {
      int row0 = mblk + wm * 32 + mt * 16 + ((lane >> 4) << 3);
      int col  = nblk + wn * 64 + nt * 16 + (lane & 15);
      float bv = HAS_BIAS ? bias[col] : 0.0f;
#pragma unroll
      for (int r = 0; r < 8; ++r) {
        float v = acc[mt][nt][r] + bv;
        if (RELU) v = fmaxf(v, 0.0f);
        if (GATED) v *= gate;
        size_t idx = (size_t)(row0 + r) * N + col;
        if (OUT_BF16) ((bf16*)Cout)[idx] = (bf16)v;
        else          ((float*)Cout)[idx] = v;
      }
    }
  }
}

// ---------------------------------------------------------------------------
// Flash attention (one block = one (b, h, 128 query rows); 8 waves x 16 rows)
// Q,K,V,ctx are bf16 [B*S, D] with head h occupying columns h*64..h*64+63.
// ---------------------------------------------------------------------------
__global__ __launch_bounds__(256) void flash_attn_wmma(
    const bf16* __restrict__ Qg, const bf16* __restrict__ Kg,
    const bf16* __restrict__ Vg, bf16* __restrict__ ctx, int S, int D) {
  constexpr int DK = 64;
  __shared__ bf16 Qs[128 * DK];    // [qrow][feat]
  __shared__ bf16 Ks[64 * DK];     // [key][feat]   (B-frag ready for Q@K^T)
  __shared__ bf16 Vst[64 * DK];    // [feat][key]   (B-frag ready for P@V)
  __shared__ bf16 Ps[8][16 * DK];  // per-wave P staging [row][key]

  const int tid = threadIdx.x, lane = tid & 31, wid = tid >> 5;
  const int h = blockIdx.y, b = blockIdx.z;
  const int q0 = blockIdx.x * 128;
  const size_t rowbase = (size_t)b * S;
  const int hc = h * DK;

  {  // load Q tile: 128x64, 32 elems/thread
    int r = tid >> 1, c = (tid & 1) << 5;
    const bf16* g = Qg + (rowbase + q0 + r) * D + hc + c;
#pragma unroll
    for (int j = 0; j < 4; ++j)
      *(v8bf*)(Qs + r * DK + c + j * 8) = *(const v8bf*)(g + j * 8);
  }
  __syncthreads();

  v16bf qf0 = load_a_frag(Qs + (wid * 16) * DK + 0,  DK, lane);
  v16bf qf1 = load_a_frag(Qs + (wid * 16) * DK + 32, DK, lane);

  v8f cacc[4] = {};  // ctx: 16 rows x 64 feats (4 N-tiles)
  float mrun[8], lrun[8];
#pragma unroll
  for (int r = 0; r < 8; ++r) { mrun[r] = -3.0e38f; lrun[r] = 0.0f; }

  for (int kt = 0; kt < S; kt += 64) {
    __syncthreads();
    {  // K tile + transposed V tile
      int r = tid >> 2, c = (tid & 3) << 4;
      const bf16* gk = Kg + (rowbase + kt + r) * D + hc + c;
      *(v8bf*)(Ks + r * DK + c)     = *(const v8bf*)(gk);
      *(v8bf*)(Ks + r * DK + c + 8) = *(const v8bf*)(gk + 8);
      const bf16* gv = Vg + (rowbase + kt + r) * D + hc + c;
      v8bf v0 = *(const v8bf*)(gv);
      v8bf v1 = *(const v8bf*)(gv + 8);
#pragma unroll
      for (int i = 0; i < 8; ++i) {
        Vst[(c + i) * DK + r]     = v0[i];
        Vst[(c + 8 + i) * DK + r] = v1[i];
      }
    }
    __syncthreads();

    // S = (Q @ K^T) * 1/sqrt(dk) : 16 rows x 64 keys (4 tiles)
    v8f sacc[4] = {};
#pragma unroll
    for (int n = 0; n < 4; ++n) {
      v16bf bk0 = load_b_frag(Ks + (n * 16) * DK + 0,  DK, lane);
      v16bf bk1 = load_b_frag(Ks + (n * 16) * DK + 32, DK, lane);
      sacc[n] = wmma_bf16(qf0, bk0, sacc[n]);
      sacc[n] = wmma_bf16(qf1, bk1, sacc[n]);
    }
#pragma unroll
    for (int n = 0; n < 4; ++n)
#pragma unroll
      for (int r = 0; r < 8; ++r) sacc[n][r] *= 0.125f;  // 1/sqrt(64)

    // Online softmax. C-layout: lane half owns rows (lane>>4)*8 + r; a row's
    // 64 values live across one 16-lane half (4 cols/lane) -> reduce in-half.
    float cmax[8];
#pragma unroll
    for (int r = 0; r < 8; ++r) {
      float m0 = fmaxf(sacc[0][r], sacc[1][r]);
      float m1 = fmaxf(sacc[2][r], sacc[3][r]);
      cmax[r] = fmaxf(m0, m1);
    }
#pragma unroll
    for (int m = 1; m < 16; m <<= 1)
#pragma unroll
      for (int r = 0; r < 8; ++r)
        cmax[r] = fmaxf(cmax[r], __shfl_xor(cmax[r], m, 32));

    float alpha[8], csum[8];
#pragma unroll
    for (int r = 0; r < 8; ++r) {
      float mnew = fmaxf(mrun[r], cmax[r]);
      alpha[r] = __expf(mrun[r] - mnew);
      mrun[r] = mnew;
      csum[r] = 0.0f;
    }
#pragma unroll
    for (int n = 0; n < 4; ++n)
#pragma unroll
      for (int r = 0; r < 8; ++r) {
        float p = __expf(sacc[n][r] - mrun[r]);
        sacc[n][r] = p;
        csum[r] += p;
      }
#pragma unroll
    for (int m = 1; m < 16; m <<= 1)
#pragma unroll
      for (int r = 0; r < 8; ++r) csum[r] += __shfl_xor(csum[r], m, 32);
#pragma unroll
    for (int r = 0; r < 8; ++r) lrun[r] = lrun[r] * alpha[r] + csum[r];
#pragma unroll
    for (int f = 0; f < 4; ++f)
#pragma unroll
      for (int r = 0; r < 8; ++r) cacc[f][r] *= alpha[r];

    // Stage P (C-layout regs -> LDS row-major) for use as A fragments
    bf16* pw = Ps[wid];
    const int rowoff = (lane >> 4) << 3;
#pragma unroll
    for (int n = 0; n < 4; ++n)
#pragma unroll
      for (int r = 0; r < 8; ++r)
        pw[(rowoff + r) * DK + n * 16 + (lane & 15)] = (bf16)sacc[n][r];
    asm volatile("s_wait_dscnt 0x0" ::: "memory");  // wave-private turnaround

    v16bf pa0 = load_a_frag(pw + 0,  DK, lane);  // keys 0..31
    v16bf pa1 = load_a_frag(pw + 32, DK, lane);  // keys 32..63
#pragma unroll
    for (int f = 0; f < 4; ++f) {
      v16bf bv0 = load_b_frag(Vst + (f * 16) * DK + 0,  DK, lane);
      v16bf bv1 = load_b_frag(Vst + (f * 16) * DK + 32, DK, lane);
      cacc[f] = wmma_bf16(pa0, bv0, cacc[f]);
      cacc[f] = wmma_bf16(pa1, bv1, cacc[f]);
    }
  }

  // Normalize and store ctx (bf16, [B*S, D] head-strided layout)
#pragma unroll
  for (int f = 0; f < 4; ++f)
#pragma unroll
    for (int r = 0; r < 8; ++r) {
      int row = q0 + wid * 16 + ((lane >> 4) << 3) + r;
      int col = hc + f * 16 + (lane & 15);
      ctx[(rowbase + row) * D + col] = (bf16)(cacc[f][r] / lrun[r]);
    }
}

// ---------------------------------------------------------------------------
// Residual-add LayerNorm, one row (D=1024) per 256-thread block
// ---------------------------------------------------------------------------
template <bool WRITE_BF16>
__global__ __launch_bounds__(256) void layernorm_kernel(
    const float* __restrict__ a, const float* __restrict__ bres,
    const float* __restrict__ g, const float* __restrict__ beta,
    float* __restrict__ outf, bf16* __restrict__ outb, int D) {
  const int row = blockIdx.x, tid = threadIdx.x;
  const int lane = tid & 31, wid = tid >> 5;
  __shared__ float wsum[8], wsum2[8];
  __shared__ float sh_mu, sh_rstd;

  const float* pa = a + (size_t)row * D;
  const float* pb = bres + (size_t)row * D;
  float v[4];
  float s = 0.0f;
#pragma unroll
  for (int i = 0; i < 4; ++i) {
    v[i] = pa[tid + i * 256] + pb[tid + i * 256];
    s += v[i];
  }
#pragma unroll
  for (int m = 1; m < 32; m <<= 1) s += __shfl_xor(s, m, 32);
  if (lane == 0) wsum[wid] = s;
  __syncthreads();
  if (tid == 0) {
    float t = 0.0f;
    for (int i = 0; i < 8; ++i) t += wsum[i];
    sh_mu = t / (float)D;
  }
  __syncthreads();
  const float mu = sh_mu;

  float s2 = 0.0f;
#pragma unroll
  for (int i = 0; i < 4; ++i) {
    float d = v[i] - mu;
    s2 += d * d;
  }
#pragma unroll
  for (int m = 1; m < 32; m <<= 1) s2 += __shfl_xor(s2, m, 32);
  if (lane == 0) wsum2[wid] = s2;
  __syncthreads();
  if (tid == 0) {
    float t = 0.0f;
    for (int i = 0; i < 8; ++i) t += wsum2[i];
    sh_rstd = rsqrtf(t / (float)D + 1e-5f);
  }
  __syncthreads();
  const float rstd = sh_rstd;

#pragma unroll
  for (int i = 0; i < 4; ++i) {
    int c = tid + i * 256;
    float y = (v[i] - mu) * rstd * g[c] + beta[c];
    outf[(size_t)row * D + c] = y;
    if (WRITE_BF16) outb[(size_t)row * D + c] = (bf16)y;
  }
}

// ---------------------------------------------------------------------------
// Launch: converts -> QKV GEMMs -> flash attn -> Wo (gated) -> LN1 -> FFN -> LN2
// ---------------------------------------------------------------------------
extern "C" void kernel_launch(void* const* d_in, const int* in_sizes, int n_in,
                              void* d_out, int out_size, void* d_ws, size_t ws_size,
                              hipStream_t stream) {
  (void)in_sizes; (void)n_in; (void)out_size; (void)ws_size;
  constexpr int B = 2, S = 2048, D = 1024, F = 4096, H = 16;
  constexpr int M = B * S;  // 4096 token rows

  const float* x     = (const float*)d_in[0];
  const float* wq    = (const float*)d_in[1];
  const float* wk    = (const float*)d_in[2];
  const float* wv    = (const float*)d_in[3];
  const float* wo    = (const float*)d_in[4];
  const float* w1    = (const float*)d_in[5];
  const float* b1    = (const float*)d_in[6];
  const float* w2    = (const float*)d_in[7];
  const float* b2    = (const float*)d_in[8];
  const float* ln1_g = (const float*)d_in[9];
  const float* ln1_b = (const float*)d_in[10];
  const float* ln2_g = (const float*)d_in[11];
  const float* ln2_b = (const float*)d_in[12];
  const float* theta = (const float*)d_in[13];

  char* ws = (char*)d_ws;
  const size_t MEG = 1048576;
  bf16*  xb   = (bf16*)(ws + 0 * MEG);    // 8 MB
  bf16*  wqt  = (bf16*)(ws + 8 * MEG);    // 2 MB  (all weights stored [N,K])
  bf16*  wkt  = (bf16*)(ws + 10 * MEG);   // 2 MB
  bf16*  wvt  = (bf16*)(ws + 12 * MEG);   // 2 MB
  bf16*  wot  = (bf16*)(ws + 14 * MEG);   // 2 MB
  bf16*  w1t  = (bf16*)(ws + 16 * MEG);   // 8 MB  [F,D]
  bf16*  w2t  = (bf16*)(ws + 24 * MEG);   // 8 MB  [D,F]
  bf16*  qb   = (bf16*)(ws + 32 * MEG);   // 8 MB
  bf16*  kb   = (bf16*)(ws + 40 * MEG);   // 8 MB
  bf16*  vb   = (bf16*)(ws + 48 * MEG);   // 8 MB
  bf16*  ctxb = (bf16*)(ws + 56 * MEG);   // 8 MB
  bf16*  hidb = (bf16*)(ws + 32 * MEG);   // 32 MB, aliases q/k/v/ctx (dead then)
  float* attn = (float*)(ws + 64 * MEG);  // 16 MB
  float* ffn  = (float*)(ws + 64 * MEG);  // 16 MB, aliases attn (dead then)
  float* x1   = (float*)(ws + 80 * MEG);  // 16 MB
  bf16*  x1b  = (bf16*)(ws + 96 * MEG);   // 8 MB   (peak ~104 MB)

  // Activations: straight convert. Weights: transpose-convert to [N,K].
  convert_f32_bf16<<<(M * D) / 1024, 256, 0, stream>>>(x, xb, M * D);
  auto tcv = [&](const float* src, bf16* dst, int K, int N) {
    transpose_convert_f32_bf16<<<dim3(N / 32, K / 32), 256, 0, stream>>>(src, dst, K, N);
  };
  tcv(wq, wqt, D, D);
  tcv(wk, wkt, D, D);
  tcv(wv, wvt, D, D);
  tcv(wo, wot, D, D);
  tcv(w1, w1t, D, F);  // -> [F,D]
  tcv(w2, w2t, F, D);  // -> [D,F]

  // Q, K, V projections (bf16 out, [M, D] head-strided layout)
  dim3 gDD(D / 128, M / 128);
  gemm_bf16_wmma<true, false, false, false><<<gDD, 256, 0, stream>>>(
      xb, wqt, qb, nullptr, nullptr, M, D, D);
  gemm_bf16_wmma<true, false, false, false><<<gDD, 256, 0, stream>>>(
      xb, wkt, kb, nullptr, nullptr, M, D, D);
  gemm_bf16_wmma<true, false, false, false><<<gDD, 256, 0, stream>>>(
      xb, wvt, vb, nullptr, nullptr, M, D, D);

  // Flash attention -> ctx (bf16)
  dim3 gAttn(S / 128, H, B);
  flash_attn_wmma<<<gAttn, 256, 0, stream>>>(qb, kb, vb, ctxb, S, D);

  // attn_out = gate * (ctx @ Wo)   (fp32)
  gemm_bf16_wmma<false, false, false, true><<<gDD, 256, 0, stream>>>(
      ctxb, wot, attn, nullptr, theta, M, D, D);

  // x1 = LN(x + attn_out)  -> fp32 + bf16
  layernorm_kernel<true><<<M, 256, 0, stream>>>(x, attn, ln1_g, ln1_b, x1, x1b, D);

  // hidden = relu(x1 @ W1 + b1)  (bf16)
  dim3 gDF(F / 128, M / 128);
  gemm_bf16_wmma<true, true, true, false><<<gDF, 256, 0, stream>>>(
      x1b, w1t, hidb, b1, nullptr, M, F, D);

  // ffn = hidden @ W2 + b2  (fp32)
  gemm_bf16_wmma<false, true, false, false><<<gDD, 256, 0, stream>>>(
      hidb, w2t, ffn, b2, nullptr, M, D, F);

  // out = LN(x1 + ffn)
  layernorm_kernel<false><<<M, 256, 0, stream>>>(x1, ffn, ln2_g, ln2_b,
                                                 (float*)d_out, nullptr, D);
}
